// SelfAttention_13572096655523
// MI455X (gfx1250) — matrix operand order repaired and checked by reading, not compiled, not run on hardware
//
#include <hip/hip_runtime.h>
#include <hip/hip_fp16.h>

typedef __attribute__((ext_vector_type(16))) _Float16 v16h;
typedef __attribute__((ext_vector_type(8)))  _Float16 v8h;
typedef __attribute__((ext_vector_type(4)))  _Float16 v4h;
typedef __attribute__((ext_vector_type(8)))  float    v8f;

#define HEADS    16
#define HEAD_DIM 64
#define EMBED    1024
#define NBATCH   4
#define QLEN     2048
#define KLEN     2048
#define PITCH    72   // LDS pitch in halfs: 144B rows -> 16B-aligned frags, odd 16B-unit stride

__device__ __forceinline__ v16h concat8(v8h lo, v8h hi) {
    return __builtin_shufflevector(lo, hi, 0,1,2,3,4,5,6,7,8,9,10,11,12,13,14,15);
}

// ---------------------------------------------------------------------------
// Kernel A: elementwise f32 -> f16 (vector4), used for K and W_out
// ---------------------------------------------------------------------------
__global__ __launch_bounds__(256)
void cvt_f32_f16_kernel(const float* __restrict__ in, _Float16* __restrict__ out, int n4)
{
    const int i = blockIdx.x * 256 + threadIdx.x;
    if (i < n4) {
        const float4 f = ((const float4*)in)[i];
        v4h h; h[0] = (_Float16)f.x; h[1] = (_Float16)f.y;
               h[2] = (_Float16)f.z; h[3] = (_Float16)f.w;
        ((v4h*)out)[i] = h;
    }
}

// ---------------------------------------------------------------------------
// Kernel B: V f32 [n][kv][E] -> f16 transposed Vt [n*H][d][kv]  (tiled via LDS)
// grid = (N*HEADS, KLEN/64), block = 256
// ---------------------------------------------------------------------------
__global__ __launch_bounds__(256)
void cvt_v_transpose_kernel(const float* __restrict__ V, _Float16* __restrict__ Vt)
{
    __shared__ _Float16 tile[HEAD_DIM * PITCH];          // [d][kv]
    const int nh  = blockIdx.x;
    const int n   = nh / HEADS, h = nh % HEADS;
    const int kv0 = blockIdx.y * 64;
    const int tid = threadIdx.x;

    for (int e = tid; e < 64 * 64; e += 256) {           // coalesced read (d fast)
        const int d = e & 63, r = e >> 6;
        tile[d * PITCH + r] =
            (_Float16)V[((size_t)(n * KLEN + kv0 + r)) * EMBED + h * HEAD_DIM + d];
    }
    __syncthreads();
    for (int e = tid; e < 64 * 64; e += 256) {           // coalesced write (kv fast)
        const int kv = e & 63, d = e >> 6;
        Vt[((size_t)(nh * HEAD_DIM + d)) * KLEN + kv0 + kv] = tile[d * PITCH + kv];
    }
}

// ---------------------------------------------------------------------------
// Kernel C: flash attention, f16 WMMA, f32 accumulate.
// grid = (N*HEADS, QLEN/128), block = 256 (8 waves); wave -> 16 q-rows.
// ---------------------------------------------------------------------------
__global__ __launch_bounds__(256)
void attn_flash_kernel(const float*    __restrict__ Q,
                       const _Float16* __restrict__ Kh,  // [n][kv][E] f16
                       const _Float16* __restrict__ Vt,  // [nh][d][kv] f16
                       const int*      __restrict__ mask,
                       _Float16*       __restrict__ attn_out)
{
    __shared__ _Float16 ldsK[64 * PITCH];                // [kv][d]   (natural)
    __shared__ _Float16 ldsV[HEAD_DIM * PITCH];          // [d][kv]   (transposed)
    __shared__ _Float16 ldsP[8][16 * PITCH];             // per-wave P patch [row][kv]

    const int tid     = threadIdx.x;
    const int wave    = tid >> 5;
    const int lane    = tid & 31;
    const int halfsel = lane >> 4;
    const int lan15   = lane & 15;

    const int nh    = blockIdx.x;
    const int n     = nh / HEADS;
    const int h     = nh % HEADS;
    const int qbase = blockIdx.y * 128 + wave * 16;

    // ---- Q A-fragments: vectorized f32 loads + one-time convert ----
    v16h qa[2];
    {
        const float* qp = Q + ((size_t)(n * QLEN + qbase + lan15)) * EMBED + h * HEAD_DIM;
        #pragma unroll
        for (int c = 0; c < 2; ++c) {
            const int dbase = c * 32 + halfsel * 8;      // A layout: K = dbase+{0..7,16..23}
            const float4 f0 = *(const float4*)&qp[dbase];
            const float4 f1 = *(const float4*)&qp[dbase + 4];
            const float4 f2 = *(const float4*)&qp[dbase + 16];
            const float4 f3 = *(const float4*)&qp[dbase + 20];
            v16h a;
            a[0]=(_Float16)f0.x; a[1]=(_Float16)f0.y; a[2]=(_Float16)f0.z; a[3]=(_Float16)f0.w;
            a[4]=(_Float16)f1.x; a[5]=(_Float16)f1.y; a[6]=(_Float16)f1.z; a[7]=(_Float16)f1.w;
            a[8]=(_Float16)f2.x; a[9]=(_Float16)f2.y; a[10]=(_Float16)f2.z; a[11]=(_Float16)f2.w;
            a[12]=(_Float16)f3.x; a[13]=(_Float16)f3.y; a[14]=(_Float16)f3.z; a[15]=(_Float16)f3.w;
            qa[c] = a;
        }
    }

    v8f acc[4];
    #pragma unroll
    for (int s = 0; s < 4; ++s) acc[s] = (v8f){0.f,0.f,0.f,0.f,0.f,0.f,0.f,0.f};
    float mrow[8], lrow[8];
    #pragma unroll
    for (int j = 0; j < 8; ++j) { mrow[j] = -3.0e38f; lrow[j] = 0.0f; }

    const float scale = 0.03125f;                        // 1/sqrt(1024)

    for (int kt = 0; kt < KLEN; kt += 64) {
        __syncthreads();
        // ---- stage K,V tiles: pure 16B copies (2 per thread each) ----
        for (int e = tid; e < 512; e += 256) {
            const int r  = e >> 3;
            const int c8 = (e & 7) * 8;
            *(v8h*)&ldsK[r * PITCH + c8] =
                *(const v8h*)&Kh[((size_t)(n * KLEN + kt + r)) * EMBED + h * HEAD_DIM + c8];
            *(v8h*)&ldsV[r * PITCH + c8] =
                *(const v8h*)&Vt[((size_t)(nh * HEAD_DIM + r)) * KLEN + kt + c8];
        }
        // prefetch next tile while this one is consumed
        if (kt + 64 < KLEN) {
            const int r = tid >> 3, c8 = (tid & 7) * 8;
            __builtin_prefetch(&Kh[((size_t)(n * KLEN + kt + 64 + r)) * EMBED + h * HEAD_DIM + c8], 0, 1);
            __builtin_prefetch(&Vt[((size_t)(nh * HEAD_DIM + r)) * KLEN + kt + 64 + c8], 0, 1);
        }
        __syncthreads();

        // ---- S = Q K^T : contiguous b128 B-frag loads ----
        v8f sacc[4];
        #pragma unroll
        for (int sub = 0; sub < 4; ++sub) {
            v8f c = (v8f){0.f,0.f,0.f,0.f,0.f,0.f,0.f,0.f};
            const int col = sub * 16 + lan15;
            #pragma unroll
            for (int kc = 0; kc < 2; ++kc) {
                const _Float16* bp = &ldsK[col * PITCH + kc * 32 + halfsel * 8];
                const v16h b = concat8(*(const v8h*)bp, *(const v8h*)(bp + 16));
                c = __builtin_amdgcn_wmma_f32_16x16x32_f16(false, qa[kc], false, b,
                                                           (short)0, c, false, false);
            }
            sacc[sub] = c;
        }

        // ---- mask (before scale, as reference), scale, row max ----
        float tmax[8];
        #pragma unroll
        for (int j = 0; j < 8; ++j) tmax[j] = -3.0e38f;
        #pragma unroll
        for (int sub = 0; sub < 4; ++sub) {
            const int col = kt + sub * 16 + lan15;
            #pragma unroll
            for (int j = 0; j < 8; ++j) {
                const int row = qbase + j + halfsel * 8;
                float s = sacc[sub][j];
                const int m = mask[((size_t)(n * QLEN + row)) * KLEN + col];
                s = (m == 0) ? -1.0e9f : s;
                s *= scale;
                sacc[sub][j] = s;
                tmax[j] = fmaxf(tmax[j], s);
            }
        }
        #pragma unroll
        for (int j = 0; j < 8; ++j) {
            float v = tmax[j];
            v = fmaxf(v, __shfl_xor(v, 1, 32));
            v = fmaxf(v, __shfl_xor(v, 2, 32));
            v = fmaxf(v, __shfl_xor(v, 4, 32));
            v = fmaxf(v, __shfl_xor(v, 8, 32));
            tmax[j] = v;
        }

        // ---- online softmax ----
        float resc[8], tsum[8];
        #pragma unroll
        for (int j = 0; j < 8; ++j) {
            const float mnew = fmaxf(mrow[j], tmax[j]);
            resc[j] = __expf(mrow[j] - mnew);
            mrow[j] = mnew;
            tsum[j] = 0.0f;
        }
        #pragma unroll
        for (int sub = 0; sub < 4; ++sub)
            #pragma unroll
            for (int j = 0; j < 8; ++j) {
                const float p = __expf(sacc[sub][j] - mrow[j]);
                sacc[sub][j] = p;
                tsum[j] += p;
            }
        #pragma unroll
        for (int j = 0; j < 8; ++j) {
            float v = tsum[j];
            v += __shfl_xor(v, 1, 32);
            v += __shfl_xor(v, 2, 32);
            v += __shfl_xor(v, 4, 32);
            v += __shfl_xor(v, 8, 32);
            lrow[j] = lrow[j] * resc[j] + v;
        }
        #pragma unroll
        for (int sub = 0; sub < 4; ++sub)
            #pragma unroll
            for (int j = 0; j < 8; ++j)
                acc[sub][j] *= resc[j];

        // ---- P: C-layout -> A-fragment via per-wave LDS patch ----
        _Float16* pw = &ldsP[wave][0];
        #pragma unroll
        for (int sub = 0; sub < 4; ++sub)
            #pragma unroll
            for (int j = 0; j < 8; ++j)
                pw[(j + halfsel * 8) * PITCH + sub * 16 + lan15] = (_Float16)sacc[sub][j];

        v16h pa[2];
        #pragma unroll
        for (int c = 0; c < 2; ++c) {
            const _Float16* pp = &ldsP[wave][lan15 * PITCH + c * 32 + halfsel * 8];
            pa[c] = concat8(*(const v8h*)pp, *(const v8h*)(pp + 16));
        }

        // ---- O += P V : contiguous b128 B-frag loads from transposed V ----
        #pragma unroll
        for (int sub = 0; sub < 4; ++sub) {
            v8f c = acc[sub];
            const int dcol = sub * 16 + lan15;
            #pragma unroll
            for (int kc = 0; kc < 2; ++kc) {
                const _Float16* bp = &ldsV[dcol * PITCH + kc * 32 + halfsel * 8];
                const v16h b = concat8(*(const v8h*)bp, *(const v8h*)(bp + 16));
                c = __builtin_amdgcn_wmma_f32_16x16x32_f16(false, pa[kc], false, b,
                                                           (short)0, c, false, false);
            }
            acc[sub] = c;
        }
    }

    // ---- normalize, store f16 attention output ----
    float inv[8];
    #pragma unroll
    for (int j = 0; j < 8; ++j) inv[j] = 1.0f / lrow[j];
    #pragma unroll
    for (int sub = 0; sub < 4; ++sub) {
        const int dcol = sub * 16 + lan15;
        #pragma unroll
        for (int j = 0; j < 8; ++j) {
            const int row = qbase + j + halfsel * 8;
            attn_out[((size_t)(n * QLEN + row)) * EMBED + h * HEAD_DIM + dcol] =
                (_Float16)(acc[sub][j] * inv[j]);
        }
    }
}

// ---------------------------------------------------------------------------
// Kernel D: projection GEMM  out[M,E] = attn[M,E] @ W^T + b   (M = N*QLEN)
// grid = (M/128, EMBED/64), block = 256
// ---------------------------------------------------------------------------
__global__ __launch_bounds__(256)
void proj_gemm_kernel(const _Float16* __restrict__ A,
                      const _Float16* __restrict__ W,
                      const float*    __restrict__ bias,
                      float*          __restrict__ out)
{
    const int tid     = threadIdx.x;
    const int wave    = tid >> 5;
    const int lane    = tid & 31;
    const int halfsel = lane >> 4;
    const int lan15   = lane & 15;

    const int mbase = blockIdx.x * 128 + wave * 16;
    const int nbase = blockIdx.y * 64;

    v8f acc[4];
    #pragma unroll
    for (int s = 0; s < 4; ++s) acc[s] = (v8f){0.f,0.f,0.f,0.f,0.f,0.f,0.f,0.f};

    const _Float16* arow = A + (size_t)(mbase + lan15) * EMBED + halfsel * 8;

    for (int kb = 0; kb < EMBED; kb += 32) {
        const _Float16* ap = arow + kb;
        const v16h a = concat8(*(const v8h*)ap, *(const v8h*)(ap + 16));

        #pragma unroll
        for (int sub = 0; sub < 4; ++sub) {
            const _Float16* bp = W + (size_t)(nbase + sub * 16 + lan15) * EMBED
                                   + kb + halfsel * 8;
            const v16h b = concat8(*(const v8h*)bp, *(const v8h*)(bp + 16));
            acc[sub] = __builtin_amdgcn_wmma_f32_16x16x32_f16(false, a, false, b,
                                                              (short)0, acc[sub],
                                                              false, false);
        }
    }

    #pragma unroll
    for (int sub = 0; sub < 4; ++sub) {
        const int nc = nbase + sub * 16 + lan15;
        const float bv = bias[nc];
        #pragma unroll
        for (int j = 0; j < 8; ++j) {
            const int mr = mbase + j + halfsel * 8;
            out[(size_t)mr * EMBED + nc] = acc[sub][j] + bv;
        }
    }
}

// ---------------------------------------------------------------------------
extern "C" void kernel_launch(void* const* d_in, const int* in_sizes, int n_in,
                              void* d_out, int out_size, void* d_ws, size_t ws_size,
                              hipStream_t stream)
{
    const float* Q    = (const float*)d_in[0];
    const float* K    = (const float*)d_in[1];
    const float* V    = (const float*)d_in[2];
    const int*   M    = (const int*)  d_in[3];
    const float* W    = (const float*)d_in[4];
    const float* bias = (const float*)d_in[5];
    float* out = (float*)d_out;

    const size_t QKV = (size_t)NBATCH * QLEN * EMBED;    // 8M elements
    _Float16* Kh     = (_Float16*)d_ws;                  // 16 MB
    _Float16* Vt     = Kh + QKV;                         // 16 MB
    _Float16* attn_h = Vt + QKV;                         // 16 MB
    _Float16* Wh     = attn_h + QKV;                     // 2 MB

    cvt_f32_f16_kernel<<<(int)(QKV / 4 / 256), 256, 0, stream>>>(K, Kh, (int)(QKV / 4));
    cvt_v_transpose_kernel<<<dim3(NBATCH * HEADS, KLEN / 64), 256, 0, stream>>>(V, Vt);
    cvt_f32_f16_kernel<<<(EMBED * EMBED / 4) / 256, 256, 0, stream>>>(W, Wh, EMBED * EMBED / 4);

    attn_flash_kernel<<<dim3(NBATCH * HEADS, QLEN / 128), 256, 0, stream>>>(Q, Kh, Vt, M, attn_h);
    proj_gemm_kernel<<<dim3((NBATCH * QLEN) / 128, EMBED / 64), 256, 0, stream>>>(attn_h, Wh, bias, out);
}
